// Model_25357486916140
// MI455X (gfx1250) — matrix operand order repaired and checked by reading, not compiled, not run on hardware
//
#include <hip/hip_runtime.h>
#include <cstdint>
#include <cstddef>

// Problem constants (match reference)
#define B_    512
#define L_    128
#define D_    256
#define DFF_  512
#define E_    8

// LDS strides padded (+8 f16 = 16B) to rotate banks between rows while
// keeping 16B alignment for ds_load_b128.
#define MT          64     // rows (tokens) per workgroup
#define XS_STRIDE   264    // 256 + 8
#define HS_STRIDE   520    // 512 + 8
#define BS1_STRIDE  264
#define BS2_STRIDE  520

// Scheduling fence: keep hand-written software pipeline shape (stop the
// machine scheduler from sinking prefetch loads back to their consumers).
#if defined(__has_builtin)
#  if __has_builtin(__builtin_amdgcn_sched_barrier)
#    define SCHED_FENCE() __builtin_amdgcn_sched_barrier(0)
#  endif
#endif
#ifndef SCHED_FENCE
#  define SCHED_FENCE()
#endif

typedef _Float16 v16h __attribute__((ext_vector_type(16)));
typedef _Float16 v8h  __attribute__((ext_vector_type(8)));
typedef _Float16 v4h  __attribute__((ext_vector_type(4)));
typedef float    v8f  __attribute__((ext_vector_type(8)));

static __device__ __forceinline__ v8f zero_v8f() {
  v8f z = {0.f,0.f,0.f,0.f,0.f,0.f,0.f,0.f};
  return z;
}

// D = A(16x32) * B(32x16) + C, f16 in / f32 accumulate
static __device__ __forceinline__ v8f wmma_f16(v16h a, v16h b, v8f c) {
  return __builtin_amdgcn_wmma_f32_16x16x32_f16(false, a, false, b,
                                                (short)0, c, false, false);
}

// A fragment (16x32 f16, M x K): lane row M = lane%16, two b128 at
// k0 = kbase + 8*half and k0+16.
static __device__ __forceinline__ v16h load_frag_a(const _Float16* p, int stride,
                                                   int row, int kbase, int half) {
  const _Float16* q = p + row * stride + kbase + half * 8;
  v8h lo = *(const v8h*)q;
  v8h hi = *(const v8h*)(q + 16);
  return __builtin_shufflevector(lo, hi, 0,1,2,3,4,5,6,7,8,9,10,11,12,13,14,15);
}

// B fragment (32x16 f16, K x N): lane col N = lane%16, two b128 at
// k0 = kbase + 8*half and k0+16 (weights staged K-contiguous per N row).
static __device__ __forceinline__ v16h load_frag_b(const _Float16* p, int stride,
                                                   int nrow, int kbase, int half) {
  const _Float16* q = p + nrow * stride + kbase + half * 8;
  v8h lo = *(const v8h*)q;
  v8h hi = *(const v8h*)(q + 16);
  return __builtin_shufflevector(lo, hi, 0,1,2,3,4,5,6,7,8,9,10,11,12,13,14,15);
}

// -------- Kernel 1: masked softmax gates (matches reference incl. EPS) -----
__global__ void moe_gates_kernel(const float* __restrict__ logits,
                                 const int* __restrict__ mask,
                                 float* __restrict__ gates) {
  int b = blockIdx.x * blockDim.x + threadIdx.x;
  if (b >= B_) return;
  float v[E_];
  float mx = -3.4e38f;
#pragma unroll
  for (int e = 0; e < E_; ++e) { v[e] = logits[b * E_ + e]; mx = fmaxf(mx, v[e]); }
  float s = 0.f;
#pragma unroll
  for (int e = 0; e < E_; ++e) { v[e] = expf(v[e] - mx); s += v[e]; }
  float sm = 0.f;
#pragma unroll
  for (int e = 0; e < E_; ++e) {
    v[e] = (mask[b * E_ + e] == 1) ? (v[e] / s) : 0.f;
    sm += v[e];
  }
  float inv = 1.0f / (sm + 1e-9f);
#pragma unroll
  for (int e = 0; e < E_; ++e) gates[b * E_ + e] = v[e] * inv;
}

// -------- Kernel 2: one-time weight convert+transpose to f16 ---------------
// W1 [E][D][DFF] f32 -> w1t [E][DFF][D] f16
// W2 [E][DFF][D] f32 -> w2t [E][D][DFF] f16
__global__ void moe_prep_kernel(const float* __restrict__ W1,
                                const float* __restrict__ W2,
                                _Float16* __restrict__ w1t,
                                _Float16* __restrict__ w2t) {
  int i = blockIdx.x * blockDim.x + threadIdx.x;   // 0 .. E*D*DFF-1 (exact grid)
  {
    int e = i >> 17;            // D*DFF = 131072
    int d = (i >> 9) & 255;     // DFF = 512
    int f = i & 511;
    w1t[(e << 17) + (f << 8) + d] = (_Float16)W1[i];
  }
  {
    int e = i >> 17;
    int f = (i >> 8) & 511;     // D = 256
    int d = i & 255;
    w2t[(e << 17) + (d << 9) + f] = (_Float16)W2[i];
  }
}

// -------- Kernel 3: fused MoE MLP (WMMA f16, f32 accumulate) ---------------
// GEMM1: A fragments register-resident -> 2 ds/WMMA.
// GEMM2: 2 output tiles share each A fragment -> 3 ds/WMMA.
// Inner loops software-pipelined one K-step ahead; staging batched 8-deep.
__global__ __launch_bounds__(256, 1)
void moe_main_kernel(const float* __restrict__ x,
                     const float* __restrict__ gates,
                     const _Float16* __restrict__ w1t,
                     const _Float16* __restrict__ w2t,
                     const float* __restrict__ b1,
                     const float* __restrict__ b2,
                     float* __restrict__ out) {
  // pool serves, in time order: (1) x tile staging (until xsf preload),
  // (2) GEMM1 B chunks 64x256 f16, (3) GEMM2 B chunks 64x512 f16.
  __shared__ _Float16 pool[64 * BS2_STRIDE];  // 66,560 B
  __shared__ _Float16 hs[MT * HS_STRIDE];     // 66,560 B

  const int t    = threadIdx.x;
  const int wave = t >> 5;
  const int lane = t & 31;
  const int half = lane >> 4;
  const int ln   = lane & 15;

  const int bid = blockIdx.x;
  const int b   = bid >> 1;          // sample
  const int m0  = (bid & 1) * MT;    // token-row offset within L
  const int m   = wave & 3;          // wave's 16-row M subtile (of 4)
  const int s   = wave >> 2;         // wave handles N-subtiles s and s+2

  // ---- stage x tile into pool: 64 x 256 f32 -> f16, batched loads ----
  {
    const float* xb = x + ((size_t)b * L_ + m0) * D_;
#pragma unroll
    for (int g = 0; g < 2; ++g) {
      float4 tmp[8];
#pragma unroll
      for (int it = 0; it < 8; ++it) {
        int flat = (g * 8 + it) * 1024 + t * 4;
        tmp[it] = *(const float4*)(xb + (flat >> 8) * D_ + (flat & 255));
      }
      SCHED_FENCE();
#pragma unroll
      for (int it = 0; it < 8; ++it) {
        int flat = (g * 8 + it) * 1024 + t * 4;
        v4h h;
        h[0] = (_Float16)tmp[it].x; h[1] = (_Float16)tmp[it].y;
        h[2] = (_Float16)tmp[it].z; h[3] = (_Float16)tmp[it].w;
        *(v4h*)(pool + (flat >> 8) * XS_STRIDE + (flat & 255)) = h;
      }
    }
  }
  __syncthreads();

  // ---- preload GEMM1 A fragments once: reused for all chunks AND experts --
  v16h xsf[8];                       // 64 VGPRs
#pragma unroll
  for (int k = 0; k < 8; ++k)
    xsf[k] = load_frag_a(pool, XS_STRIDE, m * 16 + ln, k * 32, half);
  // (pool is dead from here until the first staging barrier below)

  // persistent gate-weighted output accumulators (8 tiles of 16x16 per wave)
  v8f acc[8];                        // 64 VGPRs
#pragma unroll
  for (int i = 0; i < 8; ++i) acc[i] = zero_v8f();

  for (int e = 0; e < E_; ++e) {
    const float gv = gates[b * E_ + e];       // uniform across block
    if (gv == 0.0f) continue;                 // uniform skip: EXEC stays full

    const _Float16* w1e = w1t + ((size_t)e << 17);
    const _Float16* w2e = w2t + ((size_t)e << 17);

    // ---- GEMM1: hs[64,512] = gelu(xs[64,256] @ W1[256,512] + b1) ----
    for (int fc = 0; fc < 8; ++fc) {
      __syncthreads();                        // WAR on pool
      // stage B: w1t rows [fc*64 .. +63], 256 K each (64x256 f16), 8-deep
      {
        uint4 tmp[8];
#pragma unroll
        for (int it = 0; it < 8; ++it) {
          int row = it * 8 + (t >> 5);
          int col = (t & 31) * 8;
          tmp[it] = *(const uint4*)(w1e + ((size_t)(fc * 64 + row) << 8) + col);
        }
        SCHED_FENCE();
#pragma unroll
        for (int it = 0; it < 8; ++it) {
          int row = it * 8 + (t >> 5);
          int col = (t & 31) * 8;
          *(uint4*)(pool + row * BS1_STRIDE + col) = tmp[it];
        }
      }
      __syncthreads();

      const int f0 = fc * 64 + s * 16 + ln;         // subtile s
      const int f1 = fc * 64 + (s + 2) * 16 + ln;   // subtile s+2
      const float bias0 = b1[e * DFF_ + f0];
      const float bias1 = b1[e * DFF_ + f1];
      const int r0 = s * 16 + ln, r1 = (s + 2) * 16 + ln;
      v8f c0 = zero_v8f(), c1 = zero_v8f();
      // software-pipelined: B frags for step k+1 issued before WMMAs of k
      v16h b0c = load_frag_b(pool, BS1_STRIDE, r0, 0, half);
      v16h b1c = load_frag_b(pool, BS1_STRIDE, r1, 0, half);
#pragma unroll
      for (int k = 0; k < 8; ++k) {           // K = 256 in steps of 32
        v16h b0n, b1n;
        if (k < 7) {
          b0n = load_frag_b(pool, BS1_STRIDE, r0, (k + 1) * 32, half);
          b1n = load_frag_b(pool, BS1_STRIDE, r1, (k + 1) * 32, half);
        }
        SCHED_FENCE();
        c0 = wmma_f16(xsf[k], b0c, c0);
        c1 = wmma_f16(xsf[k], b1c, c1);
        if (k < 7) { b0c = b0n; b1c = b1n; }
      }
      // bias + exact GELU, store f16 to hs. C layout: M = v + 8*half, N = ln.
#pragma unroll
      for (int v = 0; v < 8; ++v) {
        float h0 = c0[v] + bias0;
        h0 = 0.5f * h0 * (1.0f + erff(h0 * 0.70710678118654752f));
        hs[(m * 16 + v + 8 * half) * HS_STRIDE + f0] = (_Float16)h0;
        float h1 = c1[v] + bias1;
        h1 = 0.5f * h1 * (1.0f + erff(h1 * 0.70710678118654752f));
        hs[(m * 16 + v + 8 * half) * HS_STRIDE + f1] = (_Float16)h1;
      }
    }

    // ---- GEMM2: acc += gv * (hs[64,512] @ W2[512,256] + b2) ----
#pragma unroll
    for (int dc = 0; dc < 4; ++dc) {
      __syncthreads();                        // pool WAR + hs fully written
      // stage B: w2t rows [dc*64 .. +63], 512 K each (64x512 f16), 2x8-deep
#pragma unroll
      for (int g = 0; g < 2; ++g) {
        uint4 tmp[8];
#pragma unroll
        for (int it = 0; it < 8; ++it) {
          int row = (g * 8 + it) * 4 + (t >> 6);
          int col = (t & 63) * 8;
          tmp[it] = *(const uint4*)(w2e + ((size_t)(dc * 64 + row) << 9) + col);
        }
        SCHED_FENCE();
#pragma unroll
        for (int it = 0; it < 8; ++it) {
          int row = (g * 8 + it) * 4 + (t >> 6);
          int col = (t & 63) * 8;
          *(uint4*)(pool + row * BS2_STRIDE + col) = tmp[it];
        }
      }
      __syncthreads();

      const int d0 = dc * 64 + s * 16 + ln;
      const int d1 = dc * 64 + (s + 2) * 16 + ln;
      const float bias20 = b2[e * D_ + d0];
      const float bias21 = b2[e * D_ + d1];
      const int arow = m * 16 + ln;
      const int r0 = s * 16 + ln, r1 = (s + 2) * 16 + ln;
      v8f c0 = zero_v8f(), c1 = zero_v8f();
      // software-pipelined: A+B frags for step k+1 issued before WMMAs of k
      v16h ac  = load_frag_a(hs,   HS_STRIDE,  arow, 0, half);
      v16h b0c = load_frag_b(pool, BS2_STRIDE, r0,   0, half);
      v16h b1c = load_frag_b(pool, BS2_STRIDE, r1,   0, half);
#pragma unroll
      for (int k = 0; k < 16; ++k) {          // K = 512 in steps of 32
        v16h an, b0n, b1n;
        if (k < 15) {
          an  = load_frag_a(hs,   HS_STRIDE,  arow, (k + 1) * 32, half);
          b0n = load_frag_b(pool, BS2_STRIDE, r0,   (k + 1) * 32, half);
          b1n = load_frag_b(pool, BS2_STRIDE, r1,   (k + 1) * 32, half);
        }
        SCHED_FENCE();
        c0 = wmma_f16(ac, b0c, c0);
        c1 = wmma_f16(ac, b1c, c1);
        if (k < 15) { ac = an; b0c = b0n; b1c = b1n; }
      }
#pragma unroll
      for (int v = 0; v < 8; ++v) {
        acc[2 * dc + 0][v] += gv * (c0[v] + bias20);
        acc[2 * dc + 1][v] += gv * (c1[v] + bias21);
      }
    }
  }

  // ---- write output tile (always: covers poisoned d_out even if all gates 0)
  float* ob = out + ((size_t)b * L_ + m0) * D_;
#pragma unroll
  for (int dc = 0; dc < 4; ++dc) {
    const int d0 = dc * 64 + s * 16 + ln;
    const int d1 = dc * 64 + (s + 2) * 16 + ln;
#pragma unroll
    for (int v = 0; v < 8; ++v) {
      ob[(m * 16 + v + 8 * half) * D_ + d0] = acc[2 * dc + 0][v];
      ob[(m * 16 + v + 8 * half) * D_ + d1] = acc[2 * dc + 1][v];
    }
  }
}

// ---------------------------------------------------------------------------
extern "C" void kernel_launch(void* const* d_in, const int* in_sizes, int n_in,
                              void* d_out, int out_size, void* d_ws, size_t ws_size,
                              hipStream_t stream) {
  const float* x      = (const float*)d_in[0];   // [B, L, D]
  const float* logits = (const float*)d_in[1];   // [B, E]
  const float* W1     = (const float*)d_in[2];   // [E, D, DFF]
  const float* b1     = (const float*)d_in[3];   // [E, DFF]
  const float* W2     = (const float*)d_in[4];   // [E, DFF, D]
  const float* b2     = (const float*)d_in[5];   // [E, D]
  const int*   masks  = (const int*)d_in[6];     // [B, E]
  float* out = (float*)d_out;

  // workspace layout: gates (16KB) | w1t f16 (2MB) | w2t f16 (2MB)  => ~4.02MB
  float*    gates = (float*)d_ws;
  _Float16* w1t   = (_Float16*)((char*)d_ws + 16384);
  _Float16* w2t   = w1t + (size_t)E_ * DFF_ * D_;

  moe_gates_kernel<<<dim3(2), dim3(256), 0, stream>>>(logits, masks, gates);
  moe_prep_kernel<<<dim3((E_ * D_ * DFF_) / 256), dim3(256), 0, stream>>>(W1, W2, w1t, w2t);
  moe_main_kernel<<<dim3(B_ * (L_ / MT)), dim3(256), 0, stream>>>(
      x, gates, w1t, w2t, b1, b2, out);
  (void)in_sizes; (void)n_in; (void)out_size; (void)ws_size;
}
